// MolEncoder_73014444032193
// MI455X (gfx1250) — compile-verified, matrix operand order checked by reading, not run on hardware
//
#include <hip/hip_runtime.h>
#include <cstdint>
#include <cstddef>

// ---------------------------------------------------------------------------
// MolEncoder (4x TransformerConv + pool + MLP) for gfx1250 (MI455X, wave32).
// Dense GEMMs: v_wmma_f32_16x16x32_f16 fed by double-buffered
// GLOBAL_LOAD_ASYNC_TO_LDS_B128 DMA (ASYNCcnt): tile t+1 streams into the
// alternate LDS buffer while tile t is consumed by the WMMAs.
// ---------------------------------------------------------------------------

typedef __attribute__((ext_vector_type(16))) _Float16 v16h;
typedef __attribute__((ext_vector_type(8)))  float    v8f;

#define N_NODES 50000
#define N_EDGES 200000
#define N_GRAPH 2000
#define HDIM    512     // H*D
#define DCH     128     // per-head / per-layer channel dim
#define NLAYERS 4

#define BM 64
#define BN 128
#define BK 32
#define LDT 40          // padded LDS row stride in halfs (80B rows, 16B-aligned)
#define ABYTES (BM * LDT * 2)
#define BBYTES (BN * LDT * 2)

// ---------------- float atomic max (int ordering trick) ---------------------
__device__ __forceinline__ void atomicMaxF32(float* addr, float val) {
  if (__float_as_int(val) >= 0) {
    atomicMax((int*)addr, __float_as_int(val));
  } else {
    atomicMin((unsigned int*)addr, (unsigned int)__float_as_int(val));
  }
}

// ---------------- WMMA fragment gather from LDS ------------------------------
// 16-bit A-matrix 16x32 layout (ISA 7.12.2); B staged transposed so the same
// gather yields the B fragment. Lowers to two ds_load_b128 per fragment.
__device__ __forceinline__ v16h ldsfrag(const _Float16* base, int lane) {
  int i  = lane & 15;
  int k0 = (lane >> 4) << 3;                 // 0 or 8
  const _Float16* p = base + i * LDT;
  union { v16h v; unsigned int u[8]; } f;
#pragma unroll
  for (int j = 0; j < 4; ++j) {
    f.u[j]     = *reinterpret_cast<const unsigned int*>(p + k0 + 2 * j);
    f.u[4 + j] = *reinterpret_cast<const unsigned int*>(p + 16 + k0 + 2 * j);
  }
  return f.v;
}

// 16B async DMA slots per thread per tile: one for A, two for B (rows ar, ar+64)
__device__ __forceinline__ void async_tile(const _Float16* Ap, const _Float16* Bp0,
                                           const _Float16* Bp1, unsigned la,
                                           unsigned lb0, unsigned lb1) {
  asm volatile("global_load_async_to_lds_b128 %0, %1, off" :: "v"(la),  "v"(Ap)  : "memory");
  asm volatile("global_load_async_to_lds_b128 %0, %1, off" :: "v"(lb0), "v"(Bp0) : "memory");
  asm volatile("global_load_async_to_lds_b128 %0, %1, off" :: "v"(lb1), "v"(Bp1) : "memory");
}

// ---------------- generic WMMA GEMM: C = act(A@Bt^T + bias) ------------------
// A: MxK f16 row-major. Bt: NcxK f16 row-major (pre-transposed weights).
// K%32==0, Nc%128==0. act: 0=none 1=relu 2=celu. out_half: 0 f32 / 1 f16 out.
__global__ __launch_bounds__(256)
void gemm_wmma_kernel(const _Float16* __restrict__ A, const _Float16* __restrict__ Bt,
                      const float* __restrict__ bias, void* __restrict__ Cout,
                      int M, int K, int Nc, int act, int out_half) {
  __shared__ __align__(16) _Float16 As[2 * BM * LDT];   // double-buffered
  __shared__ __align__(16) _Float16 Bs[2 * BN * LDT];

  const int tid  = threadIdx.x;
  const int lane = tid & 31;
  const int wave = tid >> 5;                 // 8 waves
  const int wm   = wave & 3;                 // 16-row tile
  const int wn   = wave >> 2;                // 64-col tile
  const int rowBase = blockIdx.y * BM;
  const int colBase = blockIdx.x * BN;

  // LDS byte offsets of the staging buffers (generic LDS ptr low 32 bits)
  const unsigned asBase = (unsigned)(size_t)(void*)As;
  const unsigned bsBase = (unsigned)(size_t)(void*)Bs;

  // per-thread async-copy slots: 16B (8 halfs) each.
  // A: row ar (0..63), chunk ac. B: rows ar and ar+64, same chunk.
  const int ar = tid >> 2;                   // 0..63
  const int ac = (tid & 3) << 3;             // 0,8,16,24 halfs
  const int agr = rowBase + ar;
  const size_t arow  = (size_t)(agr < M ? agr : M - 1) * K;  // clamped edge rows
  const size_t brow0 = (size_t)(colBase + ar) * K;
  const size_t brow1 = brow0 + (size_t)64 * K;
  const unsigned slot = (unsigned)(ar * LDT + ac) * 2u;      // byte slot in tile

  v8f acc[4] = {{}, {}, {}, {}};

  // prologue: stream tile 0 into buffer 0
  async_tile(A + arow + ac, Bt + brow0 + ac, Bt + brow1 + ac,
             asBase + slot, bsBase + slot, bsBase + slot + 64 * LDT * 2);

  int par = 0;
  for (int k0 = 0; k0 < K; k0 += BK) {
    const int kn = k0 + BK;
    if (kn < K) {
      // stream tile t+1 into the alternate buffer, overlapping tile t compute
      const unsigned np = par ^ 1;
      async_tile(A + arow + kn + ac, Bt + brow0 + kn + ac, Bt + brow1 + kn + ac,
                 asBase + np * ABYTES + slot,
                 bsBase + np * BBYTES + slot,
                 bsBase + np * BBYTES + slot + 64 * LDT * 2);
      // in-order async completion: <=3 outstanding => tile t's 3 copies done
      asm volatile("s_wait_asynccnt 0x3" ::: "memory");
    } else {
      asm volatile("s_wait_asynccnt 0x0" ::: "memory");
    }
    // L2 prefetch two tiles ahead (DMA already covers t+1); clamped, branch-free
    int kp = (k0 + 2 * BK < K) ? (k0 + 2 * BK) : k0;
    __builtin_prefetch(A + arow + kp + ac, 0, 1);
    __builtin_prefetch(Bt + brow0 + kp + ac, 0, 1);

    __syncthreads();                         // everyone's tile-t copies visible

    // ---- 4 WMMAs per wave per K step ----
    const _Float16* Ab = As + par * (BM * LDT);
    const _Float16* Bb = Bs + par * (BN * LDT);
    v16h a = ldsfrag(Ab + (16 * wm) * LDT, lane);
#pragma unroll
    for (int j = 0; j < 4; ++j) {
      v16h b = ldsfrag(Bb + (64 * wn + 16 * j) * LDT, lane);
      acc[j] = __builtin_amdgcn_wmma_f32_16x16x32_f16(false, a, false, b,
                                                      (short)0, acc[j],
                                                      false, false);
    }
    __syncthreads();                         // tile-t readers done: buffer reusable
    par ^= 1;
  }

  // C/D layout: VGPR r -> M = r (lanes 0..15) or r+8 (lanes 16..31); N = lane&15
  const int hi  = lane >> 4;
  const int nlo = lane & 15;
#pragma unroll
  for (int j = 0; j < 4; ++j) {
    const int gc = colBase + 64 * wn + 16 * j + nlo;
    const float bv = bias ? bias[gc] : 0.0f;
#pragma unroll
    for (int r = 0; r < 8; ++r) {
      int grow = rowBase + 16 * wm + r + 8 * hi;
      if (grow >= M) continue;
      float v0 = acc[j][r] + bv;
      if (act == 1) {
        v0 = fmaxf(v0, 0.0f);
      } else if (act == 2) {                 // celu(alpha=1)
        v0 = (v0 > 0.0f) ? v0 : (expf(v0) - 1.0f);
      }
      if (out_half) ((_Float16*)Cout)[(size_t)grow * Nc + gc] = (_Float16)v0;
      else          ((float*)Cout)[(size_t)grow * Nc + gc]    = v0;
    }
  }
}

// ---------------- weight transpose + f32->f16 (once per call) ----------------
// W: KxN f32 -> Wt: NxK f16.  K, N multiples of 32.
__global__ __launch_bounds__(256)
void wtrans_kernel(const float* __restrict__ W, _Float16* __restrict__ Wt,
                   int K, int N) {
  __shared__ float t[32][33];
  const int bk = blockIdx.y * 32, bn = blockIdx.x * 32;
  const int tx = threadIdx.x & 31, ty = threadIdx.x >> 5;   // 32x8
#pragma unroll
  for (int i = 0; i < 32; i += 8)
    t[ty + i][tx] = W[(size_t)(bk + ty + i) * N + bn + tx];
  __syncthreads();
#pragma unroll
  for (int i = 0; i < 32; i += 8)
    Wt[(size_t)(bn + ty + i) * K + bk + tx] = (_Float16)t[tx][ty + i];
}

// ---------------- f32 -> f16 convert (n % 4 == 0) ----------------------------
__global__ void cvt_h_kernel(const float* __restrict__ s, _Float16* __restrict__ d,
                             int n) {
  int i = (blockIdx.x * blockDim.x + threadIdx.x) * 4;
  if (i >= n) return;
  float4 v = *(const float4*)(s + i);
  union { _Float16 h[4]; uint2 u; } p;
  p.h[0] = (_Float16)v.x; p.h[1] = (_Float16)v.y;
  p.h[2] = (_Float16)v.z; p.h[3] = (_Float16)v.w;
  *(uint2*)(d + i) = p.u;
}

// ---------------- attention: alpha[e][h] = <q[dst], k[src]+e> / sqrt(D) ------
__global__ __launch_bounds__(256)
void alpha_kernel(const _Float16* __restrict__ q, const _Float16* __restrict__ k,
                  const _Float16* __restrict__ eemb, const int* __restrict__ ei,
                  float* __restrict__ alpha, int E) {
  int ed = blockIdx.x * 8 + (threadIdx.x >> 5);     // one wave per edge
  if (ed >= E) return;
  int lane = threadIdx.x & 31;
  int src = ei[ed];
  int dst = ei[E + ed];
  const uint4* qp = (const uint4*)(q    + (size_t)dst * HDIM + lane * 16);
  const uint4* kp = (const uint4*)(k    + (size_t)src * HDIM + lane * 16);
  const uint4* ep = (const uint4*)(eemb + (size_t)ed  * HDIM + lane * 16);
  union H16 { uint4 q4[2]; _Float16 h[16]; } Q, Kv, Ev;
  Q.q4[0]  = qp[0]; Q.q4[1]  = qp[1];
  Kv.q4[0] = kp[0]; Kv.q4[1] = kp[1];
  Ev.q4[0] = ep[0]; Ev.q4[1] = ep[1];
  float s = 0.0f;
#pragma unroll
  for (int c = 0; c < 16; ++c)
    s += (float)Q.h[c] * ((float)Kv.h[c] + (float)Ev.h[c]);
  // reduce over the 8 lanes of each head (lanes 8h..8h+7 cover one head)
  s += __shfl_xor(s, 1, 32);
  s += __shfl_xor(s, 2, 32);
  s += __shfl_xor(s, 4, 32);
  if ((lane & 7) == 0)
    alpha[(size_t)ed * 4 + (lane >> 3)] = s * 0.08838834764831845f; // 1/sqrt(128)
}

__global__ void segmax_kernel(const float* __restrict__ alpha,
                              const int* __restrict__ ei,
                              float* __restrict__ m, int E) {
  int i = blockIdx.x * blockDim.x + threadIdx.x;
  if (i >= E * 4) return;
  int ed = i >> 2, h = i & 3;
  int dst = ei[E + ed];
  atomicMaxF32(&m[dst * 4 + h], alpha[i]);
}

__global__ void expsum_kernel(const float* __restrict__ alpha,
                              const int* __restrict__ ei,
                              const float* __restrict__ m,
                              float* __restrict__ w, float* __restrict__ denom,
                              int E) {
  int i = blockIdx.x * blockDim.x + threadIdx.x;
  if (i >= E * 4) return;
  int ed = i >> 2, h = i & 3;
  int dst = ei[E + ed];
  float ex = expf(alpha[i] - m[dst * 4 + h]);
  w[i] = ex;
  atomicAdd(&denom[dst * 4 + h], ex);
}

// out[dst] += (ex/denom[dst]) * (v[src] + e); out pre-loaded with skip branch
__global__ __launch_bounds__(256)
void scatter_kernel(const _Float16* __restrict__ v, const _Float16* __restrict__ eemb,
                    const float* __restrict__ w, const float* __restrict__ denom,
                    const int* __restrict__ ei, float* __restrict__ out, int E) {
  int ed = blockIdx.x;
  int t  = threadIdx.x;
  int src = ei[ed];
  int dst = ei[E + ed];
  int c = t * 2;                     // c even -> c, c+1 in same head
  int h = c >> 7;
  float wh = w[(size_t)ed * 4 + h] / denom[dst * 4 + h];
  union H2 { unsigned int u; _Float16 h2[2]; } V, Ee;
  V.u  = *(const unsigned int*)(v    + (size_t)src * HDIM + c);
  Ee.u = *(const unsigned int*)(eemb + (size_t)ed  * HDIM + c);
  atomicAdd(&out[(size_t)dst * HDIM + c],     wh * ((float)V.h2[0] + (float)Ee.h2[0]));
  atomicAdd(&out[(size_t)dst * HDIM + c + 1], wh * ((float)V.h2[1] + (float)Ee.h2[1]));
}

// ---------------- pooling ----------------------------------------------------
__global__ void fill_kernel(float* p, float v, int n) {
  int i = blockIdx.x * blockDim.x + threadIdx.x;
  if (i < n) p[i] = v;
}

__global__ void initz_kernel(float* z, int n) {     // [max(128) | sum(128)] x4
  int i = blockIdx.x * blockDim.x + threadIdx.x;
  if (i >= n) return;
  int c = i & 255;
  z[i] = (c < 128) ? -__builtin_huge_valf() : 0.0f;
}

__global__ void count_kernel(const int* __restrict__ batch, float* __restrict__ cnt, int n) {
  int i = blockIdx.x * blockDim.x + threadIdx.x;
  if (i < n) atomicAdd(&cnt[batch[i]], 1.0f);
}

__global__ void pool_kernel(const float* __restrict__ h, const int* __restrict__ batch,
                            float* __restrict__ z, int layer, int n) {
  int i = blockIdx.x * blockDim.x + threadIdx.x;
  if (i >= n * DCH) return;
  int nd = i >> 7, c = i & 127;
  int g = batch[nd];
  float val = h[i];
  atomicMaxF32(&z[(size_t)g * 1024 + layer * 256 + c], val);
  atomicAdd(&z[(size_t)g * 1024 + layer * 256 + 128 + c], val);
}

__global__ void meandiv_kernel(float* __restrict__ z, const float* __restrict__ cnt, int n) {
  int i = blockIdx.x * blockDim.x + threadIdx.x;   // over G * 4 * 128 mean slots
  if (i >= n) return;
  int g = i >> 9;
  int r = i & 511;
  int l = r >> 7, c = r & 127;
  z[(size_t)g * 1024 + l * 256 + 128 + c] /= fmaxf(cnt[g], 1.0f);
}

// ---------------- final 512 -> 1 projection ----------------------------------
__global__ __launch_bounds__(128)
void head_kernel(const float* __restrict__ z2, const float* __restrict__ W3,
                 const float* __restrict__ b3, float* __restrict__ out) {
  __shared__ float red[128];
  int g = blockIdx.x, t = threadIdx.x;
  float s = 0.0f;
  for (int c = t; c < 512; c += 128)
    s += z2[(size_t)g * 512 + c] * W3[c];
  red[t] = s;
  __syncthreads();
  for (int off = 64; off > 0; off >>= 1) {
    if (t < off) red[t] += red[t + off];
    __syncthreads();
  }
  if (t == 0) out[g] = red[0] + b3[0];
}

// ---------------- host orchestration -----------------------------------------
static inline void launch_gemm(const _Float16* A, const _Float16* Bt,
                               const float* bias, void* C, int M, int K, int Nc,
                               int act, int oh, hipStream_t s) {
  dim3 grid(Nc / BN, (M + BM - 1) / BM);
  gemm_wmma_kernel<<<grid, dim3(256), 0, s>>>(A, Bt, bias, C, M, K, Nc, act, oh);
}

static inline void launch_wtrans(const float* W, _Float16* Wt, int K, int N,
                                 hipStream_t s) {
  wtrans_kernel<<<dim3(N / 32, K / 32), dim3(256), 0, s>>>(W, Wt, K, N);
}

static inline void launch_cvt(const float* src, _Float16* dst, int n, hipStream_t s) {
  cvt_h_kernel<<<(n / 4 + 255) / 256, 256, 0, s>>>(src, dst, n);
}

extern "C" void kernel_launch(void* const* d_in, const int* in_sizes, int n_in,
                              void* d_out, int out_size, void* d_ws, size_t ws_size,
                              hipStream_t stream) {
  // Inputs in setup_inputs() insertion order; params nested dicts flattened in
  // insertion order: per layer [Wq,bq,Wk,bk,Wv,bv,We,Wskip,bskip,Wlin,blin].
  const float* x     = (const float*)d_in[0];
  const int*   ei    = (const int*)d_in[1];     // [2, E]
  const float* ea    = (const float*)d_in[2];
  const int*   batch = (const int*)d_in[3];
  const float* Lp[NLAYERS][11];
  for (int l = 0; l < NLAYERS; ++l)
    for (int j = 0; j < 11; ++j)
      Lp[l][j] = (const float*)d_in[4 + l * 11 + j];
  const float* W1 = (const float*)d_in[48]; const float* b1 = (const float*)d_in[49];
  const float* W2 = (const float*)d_in[50]; const float* b2 = (const float*)d_in[51];
  const float* W3 = (const float*)d_in[52]; const float* b3 = (const float*)d_in[53];

  // workspace carve-out
  char* ws = (char*)d_ws;
  size_t off = 0;
  auto alloc = [&](size_t bytes) -> void* {
    off = (off + 255) & ~(size_t)255;
    void* p = ws + off;
    off += bytes;
    return p;
  };
  _Float16* qh    = (_Float16*)alloc((size_t)N_NODES * HDIM * 2);
  _Float16* kh    = (_Float16*)alloc((size_t)N_NODES * HDIM * 2);
  _Float16* vh    = (_Float16*)alloc((size_t)N_NODES * HDIM * 2);
  _Float16* eh    = (_Float16*)alloc((size_t)N_EDGES * HDIM * 2);
  float*    attn  = (float*)alloc((size_t)N_NODES * HDIM * 4);
  _Float16* attnh = (_Float16*)alloc((size_t)N_NODES * HDIM * 2);
  float*    hA    = (float*)alloc((size_t)N_NODES * DCH * 4);
  _Float16* hinh  = (_Float16*)alloc((size_t)N_NODES * DCH * 2);
  _Float16* eah   = (_Float16*)alloc((size_t)N_EDGES * 64 * 2);
  float* alpha    = (float*)alloc((size_t)N_EDGES * 4 * 4);
  float* wbuf     = (float*)alloc((size_t)N_EDGES * 4 * 4);
  float* m        = (float*)alloc((size_t)N_NODES * 4 * 4);
  float* denom    = (float*)alloc((size_t)N_NODES * 4 * 4);
  float* z        = (float*)alloc((size_t)N_GRAPH * 1024 * 4);
  _Float16* zh    = (_Float16*)alloc((size_t)N_GRAPH * 1024 * 2);
  _Float16* z1h   = (_Float16*)alloc((size_t)N_GRAPH * 512 * 2);
  float* z2       = (float*)alloc((size_t)N_GRAPH * 512 * 4);
  float* cnt      = (float*)alloc((size_t)N_GRAPH * 4);
  // pre-transposed f16 weights: per layer [q,k,v,skip](512x128), e(512x64),
  // lin(128x512); MLP W1t(512x1024), W2t(512x512)
  _Float16* Wt[NLAYERS][6];
  for (int l = 0; l < NLAYERS; ++l) {
    for (int j = 0; j < 4; ++j) Wt[l][j] = (_Float16*)alloc((size_t)HDIM * DCH * 2);
    Wt[l][4] = (_Float16*)alloc((size_t)HDIM * 64 * 2);
    Wt[l][5] = (_Float16*)alloc((size_t)DCH * HDIM * 2);
  }
  _Float16* W1t = (_Float16*)alloc((size_t)512 * 1024 * 2);
  _Float16* W2t = (_Float16*)alloc((size_t)512 * 512 * 2);

  const int T = 256;

  // ---- one-time per call: weight transforms + edge-attr convert + init ----
  for (int l = 0; l < NLAYERS; ++l) {
    launch_wtrans(Lp[l][0], Wt[l][0], DCH, HDIM, stream);   // Wq
    launch_wtrans(Lp[l][2], Wt[l][1], DCH, HDIM, stream);   // Wk
    launch_wtrans(Lp[l][4], Wt[l][2], DCH, HDIM, stream);   // Wv
    launch_wtrans(Lp[l][7], Wt[l][3], DCH, HDIM, stream);   // Wskip
    launch_wtrans(Lp[l][6], Wt[l][4], 64, HDIM, stream);    // We
    launch_wtrans(Lp[l][9], Wt[l][5], HDIM, DCH, stream);   // Wlin
  }
  launch_wtrans(W1, W1t, 1024, 512, stream);
  launch_wtrans(W2, W2t, 512, 512, stream);
  launch_cvt(ea, eah, N_EDGES * 64, stream);

  initz_kernel<<<(N_GRAPH * 1024 + T - 1) / T, T, 0, stream>>>(z, N_GRAPH * 1024);
  fill_kernel<<<(N_GRAPH + T - 1) / T, T, 0, stream>>>(cnt, 0.0f, N_GRAPH);
  count_kernel<<<(N_NODES + T - 1) / T, T, 0, stream>>>(batch, cnt, N_NODES);

  const float* hin = x;   // f32 current layer input (x, then hA)
  for (int l = 0; l < NLAYERS; ++l) {
    const float *bq = Lp[l][1], *bk = Lp[l][3], *bv = Lp[l][5];
    const float *bskip = Lp[l][8], *blin = Lp[l][10];

    // convert layer input once; feeds all four node projections
    launch_cvt(hin, hinh, N_NODES * DCH, stream);

    launch_gemm(hinh, Wt[l][0], bq,    qh,   N_NODES, DCH, HDIM, 0, 1, stream);
    launch_gemm(hinh, Wt[l][1], bk,    kh,   N_NODES, DCH, HDIM, 0, 1, stream);
    launch_gemm(hinh, Wt[l][2], bv,    vh,   N_NODES, DCH, HDIM, 0, 1, stream);
    launch_gemm(hinh, Wt[l][3], bskip, attn, N_NODES, DCH, HDIM, 0, 0, stream);
    launch_gemm(eah,  Wt[l][4], nullptr, eh, N_EDGES, 64,  HDIM, 0, 1, stream);

    // segment softmax state
    fill_kernel<<<(N_NODES * 4 + T - 1) / T, T, 0, stream>>>(m, -__builtin_huge_valf(), N_NODES * 4);
    fill_kernel<<<(N_NODES * 4 + T - 1) / T, T, 0, stream>>>(denom, 0.0f, N_NODES * 4);

    alpha_kernel<<<(N_EDGES + 7) / 8, 256, 0, stream>>>(qh, kh, eh, ei, alpha, N_EDGES);
    segmax_kernel<<<(N_EDGES * 4 + T - 1) / T, T, 0, stream>>>(alpha, ei, m, N_EDGES);
    expsum_kernel<<<(N_EDGES * 4 + T - 1) / T, T, 0, stream>>>(alpha, ei, m, wbuf, denom, N_EDGES);
    scatter_kernel<<<N_EDGES, 256, 0, stream>>>(vh, eh, wbuf, denom, ei, attn, N_EDGES);

    // h_next = celu(attn @ Wlin + blin)
    launch_cvt(attn, attnh, N_NODES * HDIM, stream);
    launch_gemm(attnh, Wt[l][5], blin, hA, N_NODES, HDIM, DCH, 2, 0, stream);
    pool_kernel<<<(N_NODES * DCH + T - 1) / T, T, 0, stream>>>(hA, batch, z, l, N_NODES);

    hin = hA;
  }

  meandiv_kernel<<<(N_GRAPH * 512 + T - 1) / T, T, 0, stream>>>(z, cnt, N_GRAPH * 512);

  // MLP head
  launch_cvt(z, zh, N_GRAPH * 1024, stream);
  launch_gemm(zh,  W1t, b1, z1h, N_GRAPH, 1024, 512, 1, 1, stream);
  launch_gemm(z1h, W2t, b2, z2,  N_GRAPH, 512,  512, 1, 0, stream);
  head_kernel<<<N_GRAPH, 128, 0, stream>>>(z2, W3, b3, (float*)d_out);
}
